// DotProductAttention_18605798326650
// MI455X (gfx1250) — matrix operand order repaired and checked by reading, not compiled, not run on hardware
//
#include <hip/hip_runtime.h>
#include <hip/hip_bf16.h>
#include <stdint.h>

typedef __attribute__((ext_vector_type(16))) _Float16 v16h;
typedef __attribute__((ext_vector_type(8)))  float    v8f;
typedef __attribute__((ext_vector_type(4)))  float    v4f;

#define BATCH 16
#define SEQ   2048
#define DHEAD 64
#define ROWS_PER_WG 16
#define SS_STRIDE 2052  // 2048+4: rows 16B-aligned (b128 LDS ops), 2052%64==4 spreads banks

__global__ __launch_bounds__(256)
void DotProductAttention_kernel(const float* __restrict__ q,
                                const float* __restrict__ k,
                                const float* __restrict__ v,
                                const uint8_t* __restrict__ mask,
                                float* __restrict__ out,   // [B,N,D]
                                float* __restrict__ att)   // [B,N,N]
{
    __shared__ float sS[ROWS_PER_WG * SS_STRIDE];   // score / prob stripe (f32)
    __shared__ float sRed[ROWS_PER_WG][16];
    __shared__ float sMax[ROWS_PER_WG];
    __shared__ float sInv[ROWS_PER_WG];
    __shared__ float sPart[4][ROWS_PER_WG][16];     // split-K partials for P*V

    const int b  = blockIdx.y;
    const int i0 = blockIdx.x * ROWS_PER_WG;

    const int tid    = threadIdx.x;
    const int lane   = tid & 31;
    const int wave   = tid >> 5;
    const int laneLo = lane & 15;
    const int laneHi = lane >> 4;

    const float NEG_INF = -__builtin_inff();

#define CVT4(dst, off, f) { (dst)[(off)+0] = (_Float16)(f).x; (dst)[(off)+1] = (_Float16)(f).y; \
                            (dst)[(off)+2] = (_Float16)(f).z; (dst)[(off)+3] = (_Float16)(f).w; }

    // ---- Q stripe (16x64) as two f16 A-fragments (K=0..31 and K=32..63) ----
    // 16-bit A layout: lanes 0-15 = rows 0-15 holding K {0..7,16..23},
    //                  lanes 16-31 = rows 0-15 holding K {8..15,24..31}.
    // Per lane that is contiguous float runs -> explicit float4 (global_load_b128).
    const float4* q4 = (const float4*)(q + ((size_t)b * SEQ + i0 + laneLo) * DHEAD);
    v16h a0, a1;
    {
        float4 f0 = q4[laneHi * 2 + 0];       // K =      laneHi*8 + 0..7
        float4 f1 = q4[laneHi * 2 + 1];
        float4 f2 = q4[4 + laneHi * 2 + 0];   // K = 16 + laneHi*8 + 0..7
        float4 f3 = q4[4 + laneHi * 2 + 1];
        float4 g0 = q4[8 + laneHi * 2 + 0];   // same pattern, K += 32
        float4 g1 = q4[8 + laneHi * 2 + 1];
        float4 g2 = q4[12 + laneHi * 2 + 0];
        float4 g3 = q4[12 + laneHi * 2 + 1];
        CVT4(a0, 0, f0) CVT4(a0, 4, f1) CVT4(a0, 8, f2) CVT4(a0, 12, f3)
        CVT4(a1, 0, g0) CVT4(a1, 4, g1) CVT4(a1, 8, g2) CVT4(a1, 12, g3)
    }

    // ---- Phase 1: S = Q K^T into LDS (mask applied later, coalesced) ----
    for (int jt = wave; jt < SEQ / 16; jt += 8) {
        const int n0 = jt * 16;
        const float* krow = k + ((size_t)b * SEQ + n0 + laneLo) * DHEAD;
        if (jt + 8 < SEQ / 16)
            __builtin_prefetch(krow + 128 * DHEAD, 0, 1);  // next tile for this wave
        // 16-bit B layout 32x16: lanes 0-15 hold K {0..7,16..23}, lanes 16-31 hold K {8..15,24..31}
        const float4* k4 = (const float4*)krow;
        v16h b0, b1;
        {
            float4 f0 = k4[laneHi * 2 + 0];       // K = laneHi*8 + 0..7, 16 + laneHi*8 + 0..7
            float4 f1 = k4[laneHi * 2 + 1];
            float4 f2 = k4[4 + laneHi * 2 + 0];
            float4 f3 = k4[4 + laneHi * 2 + 1];
            float4 g0 = k4[8 + laneHi * 2 + 0];   // K += 32
            float4 g1 = k4[8 + laneHi * 2 + 1];
            float4 g2 = k4[12 + laneHi * 2 + 0];
            float4 g3 = k4[12 + laneHi * 2 + 1];
            CVT4(b0, 0, f0) CVT4(b0, 4, f1) CVT4(b0, 8, f2) CVT4(b0, 12, f3)
            CVT4(b1, 0, g0) CVT4(b1, 4, g1) CVT4(b1, 8, g2) CVT4(b1, 12, g3)
        }
        v8f c = {};
        c = __builtin_amdgcn_wmma_f32_16x16x32_f16(false, a0, false, b0, (short)0, c, false, false);
        c = __builtin_amdgcn_wmma_f32_16x16x32_f16(false, a1, false, b1, (short)0, c, false, false);
        const int col = n0 + laneLo;
#pragma unroll
        for (int r = 0; r < 8; ++r)
            sS[(r + (laneHi << 3)) * SS_STRIDE + col] = c[r];  // conflict-free stores
    }
    __syncthreads();

    // ---- Phase 2a: apply mask with fully-coalesced u32 loads (NT: read-once) ----
    for (int r = 0; r < ROWS_PER_WG; ++r) {
        const uint32_t* mrow = (const uint32_t*)(mask + ((size_t)b * SEQ + (i0 + r)) * SEQ);
        float* srow = sS + r * SS_STRIDE;
#pragma unroll
        for (int it = 0; it < SEQ / 4 / 256; ++it) {
            const int c4 = tid + it * 256;
            const uint32_t mm = __builtin_nontemporal_load(mrow + c4);
            if (mm) {
                const int c = c4 * 4;
                if (mm & 0x000000FFu) srow[c + 0] = NEG_INF;
                if (mm & 0x0000FF00u) srow[c + 1] = NEG_INF;
                if (mm & 0x00FF0000u) srow[c + 2] = NEG_INF;
                if (mm & 0xFF000000u) srow[c + 3] = NEG_INF;
            }
        }
    }
    __syncthreads();

    // ---- Phase 2b: softmax stats (16 threads cooperate per row) ----
    const int r2 = tid >> 4;   // row 0..15
    const int c2 = tid & 15;
    float mx = NEG_INF;
    for (int c = c2; c < SEQ; c += 16) mx = fmaxf(mx, sS[r2 * SS_STRIDE + c]);
    sRed[r2][c2] = mx;
    __syncthreads();
    if (c2 == 0) {
        float m = sRed[r2][0];
#pragma unroll
        for (int i = 1; i < 16; ++i) m = fmaxf(m, sRed[r2][i]);
        sMax[r2] = m;
    }
    __syncthreads();
    const float rmax = sMax[r2];
    float sum = 0.f;
    for (int c = c2; c < SEQ; c += 16) {
        const float e = __expf(sS[r2 * SS_STRIDE + c] - rmax);
        sS[r2 * SS_STRIDE + c] = e;
        sum += e;
    }
    sRed[r2][c2] = sum;
    __syncthreads();
    if (c2 == 0) {
        float s = 0.f;
#pragma unroll
        for (int i = 0; i < 16; ++i) s += sRed[r2][i];
        sInv[r2] = 1.f / s;
    }
    __syncthreads();

    // ---- Phase 2c: normalize + stream probs to global.
    // Block-wide row sweep: 4 KB contiguous per iteration, non-temporal stores
    // (268 MB write-once stream: keep it out of L2 so K/V stay resident). ----
    for (int r = 0; r < ROWS_PER_WG; ++r) {
        const float rinv = sInv[r];
        v4f* attRow = (v4f*)(att + ((size_t)b * SEQ + (i0 + r)) * SEQ);
        float* srow = sS + r * SS_STRIDE;
#pragma unroll
        for (int it = 0; it < SEQ / 4 / 256; ++it) {
            const int c4 = tid + it * 256;
            const int c  = c4 * 4;
            v4f p;
            p[0] = srow[c + 0] * rinv;
            p[1] = srow[c + 1] * rinv;
            p[2] = srow[c + 2] * rinv;
            p[3] = srow[c + 3] * rinv;
            *(v4f*)(srow + c) = p;                     // ds_store_b128 (16B aligned)
            __builtin_nontemporal_store(p, attRow + c4);
        }
    }
    __syncthreads();

    // ---- Phase 3: out = P @ V (split-K across wave pairs) ----
    const int tile  = wave & 3;   // which 16-wide slice of D
    const int khalf = wave >> 2;  // K half: [0,1024) or [1024,2048)
    v8f acc = {};
    for (int kt = khalf * 32; kt < khalf * 32 + 32; ++kt) {
        const int kbase = kt * 32;
        v16h pa, vb;
#pragma unroll
        for (int h = 0; h < 16; ++h) {
            const int kk = kbase + (h & 7) + ((h >> 3) << 4) + (laneHi << 3);
            pa[h] = (_Float16)sS[laneLo * SS_STRIDE + kk];          // A: probs
            const int kd = kbase + (h & 7) + ((h >> 3) << 4) + (laneHi << 3);
            vb[h] = (_Float16)v[((size_t)b * SEQ + kd) * DHEAD + tile * 16 + laneLo]; // B: V
        }
        acc = __builtin_amdgcn_wmma_f32_16x16x32_f16(false, pa, false, vb, (short)0, acc, false, false);
    }
    if (khalf == 1) {
#pragma unroll
        for (int r = 0; r < 8; ++r)
            sPart[tile][r + (laneHi << 3)][laneLo] = acc[r];
    }
    __syncthreads();
    if (khalf == 0) {
#pragma unroll
        for (int r = 0; r < 8; ++r) {
            const int m = r + (laneHi << 3);
            const float val = acc[r] + sPart[tile][m][laneLo];
            out[((size_t)b * SEQ + (i0 + m)) * DHEAD + tile * 16 + laneLo] = val;
        }
    }
#undef CVT4
}

extern "C" void kernel_launch(void* const* d_in, const int* in_sizes, int n_in,
                              void* d_out, int out_size, void* d_ws, size_t ws_size,
                              hipStream_t stream) {
    (void)in_sizes; (void)n_in; (void)out_size; (void)d_ws; (void)ws_size;
    const float*   q    = (const float*)d_in[0];
    const float*   k    = (const float*)d_in[1];
    const float*   v    = (const float*)d_in[2];
    const uint8_t* mask = (const uint8_t*)d_in[3];   // jnp.bool_ -> 1 byte/elem
    float* out = (float*)d_out;                                  // [B,N,D] first
    float* att = out + (size_t)BATCH * SEQ * DHEAD;              // then [B,N,N]
    dim3 grid(SEQ / ROWS_PER_WG, BATCH);
    DotProductAttention_kernel<<<grid, 256, 0, stream>>>(q, k, v, mask, out, att);
}
